// MultiHeadSelfAttention_84104049590425
// MI455X (gfx1250) — matrix operand order repaired and checked by reading, not compiled, not run on hardware
//
#include <hip/hip_runtime.h>
#include <hip/hip_bf16.h>
#include <stdint.h>

typedef __attribute__((ext_vector_type(16))) __bf16 v16bf;
typedef __attribute__((ext_vector_type(8)))  __bf16 v8bf;
typedef __attribute__((ext_vector_type(8)))  float  v8f;
typedef __attribute__((ext_vector_type(4)))  int    v4i;

#define DIM 1024
#define HEADS 16
#define HEAD_DIM 64
#define BATCH 4
#define SEQ 2048
#define MROWS (BATCH * SEQ)   // 8192

// ---------- CDNA5 async global->LDS path (guarded; falls back to sync) ------
#if defined(__gfx1250__) &&                                                  \
    __has_builtin(__builtin_amdgcn_global_load_async_to_lds_b128) &&         \
    __has_builtin(__builtin_amdgcn_s_wait_asynccnt)
#define HAVE_ASYNC_LDS 1
#else
#define HAVE_ASYNC_LDS 0
#endif

#if HAVE_ASYNC_LDS
typedef __attribute__((address_space(1))) v4i gv4i;  // global int4
typedef __attribute__((address_space(3))) v4i lv4i;  // LDS int4
__device__ __forceinline__ void async_g2l_b128(const __bf16* gsrc, __bf16* ldst) {
  // per-lane 16B DMA global -> LDS, tracked by ASYNCcnt
  __builtin_amdgcn_global_load_async_to_lds_b128((gv4i*)gsrc, (lv4i*)ldst, 0, 0);
}
#endif

// ---------- helpers ----------
__device__ __forceinline__ __bf16 to_bf(float x) {
  uint32_t u = __builtin_bit_cast(uint32_t, x);
  u += 0x7FFFu + ((u >> 16) & 1u);          // round-to-nearest-even
  uint16_t hs = (uint16_t)(u >> 16);
  return __builtin_bit_cast(__bf16, hs);
}
__device__ __forceinline__ __bf16 to_bf(__bf16 x) { return x; }

// 16 contiguous bf16 -> v16bf (two 16B loads)
__device__ __forceinline__ v16bf ld16(const __bf16* p) {
  v8bf lo = *(const v8bf*)p;
  v8bf hi = *(const v8bf*)(p + 8);
  v16bf r;
#pragma unroll
  for (int i = 0; i < 8; ++i) { r[i] = lo[i]; r[i + 8] = hi[i]; }
  return r;
}
// two separate 8-element runs (A-matrix layout)
__device__ __forceinline__ v16bf ld8x2(const __bf16* p0, const __bf16* p1) {
  v8bf lo = *(const v8bf*)p0;
  v8bf hi = *(const v8bf*)p1;
  v16bf r;
#pragma unroll
  for (int i = 0; i < 8; ++i) { r[i] = lo[i]; r[i + 8] = hi[i]; }
  return r;
}
__device__ __forceinline__ v8f wmma_bf16(v16bf a, v16bf b, v8f c) {
  return __builtin_amdgcn_wmma_f32_16x16x32_bf16(false, a, false, b, (short)0, c,
                                                 false, false);
}
__device__ __forceinline__ v8f vzero8() {
  v8f z;
#pragma unroll
  for (int i = 0; i < 8; ++i) z[i] = 0.f;
  return z;
}

// ---------- fp32 -> bf16 convert ----------
__global__ void f32_to_bf16_kernel(const float* __restrict__ in,
                                   __bf16* __restrict__ out, int n) {
  int i = blockIdx.x * blockDim.x + threadIdx.x;
  if (i < n) out[i] = to_bf(in[i]);
}

// ---------- tiled WMMA GEMM: C[M,N] = A[M,K] * B[K,N] ----------
// MODE 0: scatter bf16 results into per-head Q/K/V  (N == 3*DIM)
// MODE 1: fp32 output + bias                          (N == DIM)
#define BM 128
#define BN 128
#define BK 32
#define LDSA 40   // BK + 8 pad, rows stay 16B aligned (80B stride)
#define LDSB 40   // BsT rows (n-major), 16B aligned

template <int MODE, typename TA>
__global__ __launch_bounds__(256) void gemm_wmma_kernel(
    const TA* __restrict__ A, const __bf16* __restrict__ B, int M, int N, int K,
    __bf16* __restrict__ qo, __bf16* __restrict__ ko, __bf16* __restrict__ vo,
    const float* __restrict__ bias, float* __restrict__ fout) {
  __shared__ __bf16 As[BM * LDSA];
  __shared__ __bf16 BsT[BN * LDSB];

  const int tid = threadIdx.x;
  const int m0 = blockIdx.x * BM;
  const int n0 = blockIdx.y * BN;

  const int wid = tid >> 5;
  const int lane = tid & 31;
  const int h = lane >> 4;       // lane half
  const int lm = lane & 15;
  const int wm = (wid & 1) * 64; // wave M offset in tile
  const int wn = (wid >> 1) * 32;

  v8f acc[4][2];
#pragma unroll
  for (int i = 0; i < 4; ++i)
#pragma unroll
    for (int j = 0; j < 2; ++j) acc[i][j] = vzero8();

  for (int kt = 0; kt < K; kt += BK) {
    { // stage A: 128x32 tile, 16 elems/thread, converted to bf16
      int row = tid >> 1;
      int cb = (tid & 1) << 4;
      const TA* pa = A + (size_t)(m0 + row) * K + kt + cb;
#pragma unroll
      for (int i = 0; i < 16; ++i) As[row * LDSA + cb + i] = to_bf(pa[i]);
      if (kt + BK < K)  // hide next A tile's HBM latency behind this tile's math
        __builtin_prefetch(pa + BK, 0, 1);
    }
    { // stage B transposed: BsT[n][k]
      int row = tid >> 3;       // k within tile: 0..31
      int cb = (tid & 7) << 4;  // n within tile
      const __bf16* pb = B + (size_t)(kt + row) * N + n0 + cb;
#pragma unroll
      for (int i = 0; i < 16; ++i) BsT[(cb + i) * LDSB + row] = pb[i];
      if (kt + BK < K)
        __builtin_prefetch(pb + (size_t)BK * N, 0, 1);
    }
    __syncthreads();

    v16bf af[4], bfr[2];
#pragma unroll
    for (int ms = 0; ms < 4; ++ms) {
      const __bf16* pr = &As[(wm + ms * 16 + lm) * LDSA];
      af[ms] = ld8x2(pr + h * 8, pr + 16 + h * 8);  // A layout
    }
#pragma unroll
    for (int ns = 0; ns < 2; ++ns)
      bfr[ns] = ld8x2(&BsT[(wn + ns * 16 + lm) * LDSB + h * 8], &BsT[(wn + ns * 16 + lm) * LDSB + 16 + h * 8]);  // B layout

#pragma unroll
    for (int ms = 0; ms < 4; ++ms)
#pragma unroll
      for (int ns = 0; ns < 2; ++ns)
        acc[ms][ns] = wmma_bf16(af[ms], bfr[ns], acc[ms][ns]);

    __syncthreads();
  }

  // epilogue; C layout: M = r + 8*h, N = lm
#pragma unroll
  for (int ms = 0; ms < 4; ++ms) {
#pragma unroll
    for (int ns = 0; ns < 2; ++ns) {
#pragma unroll
      for (int r = 0; r < 8; ++r) {
        int grow = m0 + wm + ms * 16 + r + 8 * h;
        int gcol = n0 + wn + ns * 16 + lm;
        float v = acc[ms][ns][r];
        if (MODE == 0) {
          int which = gcol >> 10;                 // 0=q 1=k 2=v
          int head = (gcol >> 6) & (HEADS - 1);
          int d = gcol & (HEAD_DIM - 1);
          int b = grow >> 11;
          int n = grow & (SEQ - 1);
          size_t dst = ((size_t)(b * HEADS + head) * SEQ + n) * HEAD_DIM + d;
          __bf16* o = (which == 0) ? qo : (which == 1) ? ko : vo;
          o[dst] = to_bf(v);
        } else {
          fout[(size_t)grow * N + gcol] = v + bias[gcol];
        }
      }
    }
  }
}

// ---------- flash attention over per-head Q/K/V [B*H, SEQ, 64] bf16 ----------
#define FA_LD 72  // 64 + 8 pad (144B rows, 16B aligned)

__global__ __launch_bounds__(256) void flash_attn_kernel(
    const __bf16* __restrict__ Q, const __bf16* __restrict__ Kh,
    const __bf16* __restrict__ V, __bf16* __restrict__ ctx) {
  __shared__ __bf16 Ks[64 * FA_LD];        // [key][d]
  __shared__ __bf16 VsT[64 * FA_LD];       // [d][key]
  __shared__ __bf16 Ps[8 * 16 * FA_LD];    // per-wave 16 x 64 P tile

  const int tid = threadIdx.x;
  const int w = tid >> 5;
  const int lane = tid & 31;
  const int h = lane >> 4;
  const int lm = lane & 15;

  const int bh = blockIdx.y;               // 0..63
  const int qbase = blockIdx.x * 128;
  const size_t headoff = (size_t)bh * SEQ * HEAD_DIM;
  const float scale = 0.125f;              // HEAD_DIM^-0.5

  // Q A-fragments for this wave's 16 rows (hd split into 2 K-steps of 32)
  const int qrow = qbase + w * 16 + lm;
  const __bf16* qptr = Q + headoff + (size_t)qrow * HEAD_DIM;
  v16bf qf[2];
#pragma unroll
  for (int kk = 0; kk < 2; ++kk)
    qf[kk] = ld8x2(qptr + kk * 32 + h * 8, qptr + kk * 32 + 16 + h * 8);

  v8f o[4];
#pragma unroll
  for (int ns = 0; ns < 4; ++ns) o[ns] = vzero8();
  float m[8], l[8];
#pragma unroll
  for (int r = 0; r < 8; ++r) { m[r] = -1e30f; l[r] = 0.f; }

  for (int jc = 0; jc < SEQ; jc += 64) {
    { // stage K chunk (async DMA global->LDS) and V chunk (transposed, VGPR path)
      int kr = tid >> 2;              // 0..63
      int cb = (tid & 3) << 4;        // 0,16,32,48
      const __bf16* pk = Kh + headoff + (size_t)(jc + kr) * HEAD_DIM + cb;
#if HAVE_ASYNC_LDS
      async_g2l_b128(pk, &Ks[kr * FA_LD + cb]);
      async_g2l_b128(pk + 8, &Ks[kr * FA_LD + cb + 8]);
#else
      *(v8bf*)&Ks[kr * FA_LD + cb] = *(const v8bf*)pk;
      *(v8bf*)&Ks[kr * FA_LD + cb + 8] = *(const v8bf*)(pk + 8);
#endif
      const __bf16* pv = V + headoff + (size_t)(jc + kr) * HEAD_DIM + cb;
#pragma unroll
      for (int i = 0; i < 16; ++i) VsT[(cb + i) * FA_LD + kr] = pv[i];
      if (jc + 64 < SEQ) {  // prefetch next chunk's K/V rows into L2/L0
        __builtin_prefetch(pk + (size_t)64 * HEAD_DIM, 0, 1);
        __builtin_prefetch(pv + (size_t)64 * HEAD_DIM, 0, 1);
      }
    }
#if HAVE_ASYNC_LDS
    __builtin_amdgcn_s_wait_asynccnt(0);
#endif
    __syncthreads();

    // S = Q * K^T   (B[k=d][n=key] = Ks[key][d], contiguous in d)
    v8f s[4];
#pragma unroll
    for (int ns = 0; ns < 4; ++ns) s[ns] = vzero8();
#pragma unroll
    for (int ns = 0; ns < 4; ++ns)
#pragma unroll
      for (int kk = 0; kk < 2; ++kk) {
        v16bf bfrag = ld8x2(&Ks[(ns * 16 + lm) * FA_LD + kk * 32 + h * 8], &Ks[(ns * 16 + lm) * FA_LD + kk * 32 + 16 + h * 8]);
        s[ns] = wmma_bf16(qf[kk], bfrag, s[ns]);
      }

    // online softmax; row of element r is (r + 8*h); cols span 16 lanes/half
    float cf[8];
#pragma unroll
    for (int r = 0; r < 8; ++r) {
      float cm = -1e30f;
#pragma unroll
      for (int ns = 0; ns < 4; ++ns) {
        s[ns][r] *= scale;
        cm = fmaxf(cm, s[ns][r]);
      }
#pragma unroll
      for (int off = 1; off < 16; off <<= 1)
        cm = fmaxf(cm, __shfl_xor(cm, off, 32));
      float nm = fmaxf(m[r], cm);
      cf[r] = __expf(m[r] - nm);
      m[r] = nm;
      float rs = 0.f;
#pragma unroll
      for (int ns = 0; ns < 4; ++ns) {
        float pv = __expf(s[ns][r] - nm);
        rs += pv;
        Ps[(w * 16 + r + 8 * h) * FA_LD + ns * 16 + lm] = to_bf(pv);
      }
#pragma unroll
      for (int off = 1; off < 16; off <<= 1) rs += __shfl_xor(rs, off, 32);
      l[r] = l[r] * cf[r] + rs;
#pragma unroll
      for (int ns = 0; ns < 4; ++ns) o[ns][r] *= cf[r];
    }
    __syncthreads();  // Ps visible (C-layout -> A-layout bounce)

    // O += P * V   (A from Ps, B[k=key][n=d] = VsT[d][key], contiguous in key)
    v16bf pf[2];
#pragma unroll
    for (int kk = 0; kk < 2; ++kk) {
      const __bf16* pp = &Ps[(w * 16 + lm) * FA_LD + kk * 32];
      pf[kk] = ld8x2(pp + h * 8, pp + 16 + h * 8);
    }
#pragma unroll
    for (int ns = 0; ns < 4; ++ns)
#pragma unroll
      for (int kk = 0; kk < 2; ++kk) {
        v16bf vfrag = ld8x2(&VsT[(ns * 16 + lm) * FA_LD + kk * 32 + h * 8], &VsT[(ns * 16 + lm) * FA_LD + kk * 32 + 16 + h * 8]);
        o[ns] = wmma_bf16(pf[kk], vfrag, o[ns]);
      }
    __syncthreads();  // protect Ks/VsT/Ps before next chunk
  }

  // normalize and write context, packed [B*N, DIM] with head column offset
  const int b = bh >> 4;
  const int head = bh & 15;
#pragma unroll
  for (int r = 0; r < 8; ++r) {
    float inv = 1.f / l[r];
    int q = qbase + w * 16 + r + 8 * h;
    size_t rowoff = ((size_t)(b * SEQ + q)) * DIM + head * HEAD_DIM;
#pragma unroll
    for (int ns = 0; ns < 4; ++ns)
      ctx[rowoff + ns * 16 + lm] = to_bf(o[ns][r] * inv);
  }
}

// ---------- launch ----------
extern "C" void kernel_launch(void* const* d_in, const int* in_sizes, int n_in,
                              void* d_out, int out_size, void* d_ws,
                              size_t ws_size, hipStream_t stream) {
  const float* x = (const float*)d_in[0];       // [B,N,D]
  const float* w_qkv = (const float*)d_in[1];   // [D,3D]
  const float* w_out = (const float*)d_in[2];   // [D,D]
  const float* b_out = (const float*)d_in[3];   // [D]
  float* out = (float*)d_out;                   // [B,N,D]

  char* ws = (char*)d_ws;
  __bf16* wqkvb = (__bf16*)ws; ws += (size_t)DIM * 3 * DIM * 2;                  // 6 MB
  __bf16* woutb = (__bf16*)ws; ws += (size_t)DIM * DIM * 2;                      // 2 MB
  __bf16* Qh = (__bf16*)ws;    ws += (size_t)BATCH * HEADS * SEQ * HEAD_DIM * 2; // 16 MB
  __bf16* Kb = (__bf16*)ws;    ws += (size_t)BATCH * HEADS * SEQ * HEAD_DIM * 2; // 16 MB
  __bf16* Vb = (__bf16*)ws;    ws += (size_t)BATCH * HEADS * SEQ * HEAD_DIM * 2; // 16 MB
  __bf16* ctx = (__bf16*)ws;   ws += (size_t)MROWS * DIM * 2;                    // 16 MB
  (void)ws_size; (void)in_sizes; (void)n_in; (void)out_size;

  // 1) weights to bf16
  {
    int n1 = 3 * DIM * DIM;
    f32_to_bf16_kernel<<<(n1 + 255) / 256, 256, 0, stream>>>(w_qkv, wqkvb, n1);
    int n2 = DIM * DIM;
    f32_to_bf16_kernel<<<(n2 + 255) / 256, 256, 0, stream>>>(w_out, woutb, n2);
  }
  // 2) QKV projection, scatter into per-head Q/K/V
  gemm_wmma_kernel<0, float><<<dim3(MROWS / BM, (3 * DIM) / BN), 256, 0, stream>>>(
      x, wqkvb, MROWS, 3 * DIM, DIM, Qh, Kb, Vb, nullptr, nullptr);
  // 3) flash attention -> ctx
  flash_attn_kernel<<<dim3(SEQ / 128, BATCH * HEADS), 256, 0, stream>>>(Qh, Kb, Vb, ctx);
  // 4) output projection + bias -> fp32 out
  gemm_wmma_kernel<1, __bf16><<<dim3(MROWS / BM, DIM / BN), 256, 0, stream>>>(
      ctx, woutb, MROWS, DIM, DIM, nullptr, nullptr, nullptr, b_out, out);
}
